// ModernBertAttention_64106681860810
// MI455X (gfx1250) — matrix operand order repaired and checked by reading, not compiled, not run on hardware
//
#include <hip/hip_runtime.h>

typedef __attribute__((ext_vector_type(16))) __bf16 v16bf;
typedef __attribute__((ext_vector_type(8)))  __bf16 v8bf;
typedef __attribute__((ext_vector_type(8)))  float  v8f;

#define B_    2
#define S_    2048
#define HID_  768
#define NH_   12
#define HD_   64
#define WIN_  64
#define QKVN_ (3 * HID_)        // 2304
#define SP_   (S_ + 160)        // padded V time dim: 64 front pad + 96 back pad
#define LDSW_ 40                // LDS tile row stride in elements (bank-conflict free)

static __device__ __forceinline__ v16bf cat8(v8bf a, v8bf b) {
    return __builtin_shufflevector(a, b, 0,1,2,3,4,5,6,7,8,9,10,11,12,13,14,15);
}

static __device__ __forceinline__ v8f wmma_bf16(v16bf a, v16bf b, v8f c) {
    // D = A(16x32 bf16) * B(32x16 bf16) + C(16x16 f32)
    return __builtin_amdgcn_wmma_f32_16x16x32_bf16(false, a, false, b, (short)0, c, false, false);
}

// ---------------------------------------------------------------------------
// Tensor Data Mover support (CDNA5 TDM). Guarded so compilation never fails.
// ---------------------------------------------------------------------------
#if defined(__has_builtin)
#if __has_builtin(__builtin_amdgcn_tensor_load_to_lds)
#define HAVE_TDM 1
#endif
#endif

#ifdef HAVE_TDM
typedef __attribute__((ext_vector_type(4))) unsigned int u32x4;
typedef __attribute__((ext_vector_type(8))) int          i32x8;
typedef __attribute__((ext_vector_type(4))) int          i32x4;

// Issue one TDM load of a 128-row x 32-col bf16 tile into LDS.
//   global tile base = g (row stride K elements, 2 bytes each)
//   LDS destination: row stride LDSW_(40) elements, produced by TDM padding:
//     pad_interval code 3 -> every 16 dwords (64B = one 32-elem bf16 row)
//     pad_amount  code 3 -> insert 4 dwords (16B = 8 elems)  => stride 40
// (CDNA5 ISA: D# group0/group1 bitfields)
static __device__ __forceinline__ void tdm_load_tile_128x32(
    const __bf16* g, unsigned lds_byte_addr, int tensor_rows, int K)
{
    size_t ga = (size_t)g;
    u32x4 g0;
    g0[0] = 1u;                                       // count=1 valid user D#
    g0[1] = lds_byte_addr;                            // lds_addr (bytes)
    g0[2] = (unsigned)(ga & 0xFFFFFFFFu);             // global_addr[31:0]
    g0[3] = (unsigned)((ga >> 32) & 0x01FFFFFFu)      // global_addr[56:32]
          | (2u << 30);                               // type = 2 ("image")
    i32x8 g1;
    g1[0] = (int)((1u << 16)     // data_size = 1 -> 2 bytes
                | (1u << 20)     // pad_enable
                | (3u << 22)     // pad_interval: 16 dwords
                | (3u << 25));   // pad_amount: 4 dwords
    g1[1] = (int)((unsigned)K << 16);                          // tensor_dim0 lo16
    g1[2] = (int)(((unsigned)K >> 16)
                | ((unsigned)tensor_rows << 16));              // dim0 hi | dim1 lo
    g1[3] = (int)(((unsigned)tensor_rows >> 16) | (32u << 16));// dim1 hi | tile_dim0=32
    g1[4] = 128;                                               // tile_dim1=128, tile_dim2=0
    g1[5] = K;                                                 // tensor_dim0_stride lo32
    g1[6] = 0;                                                 // stride hi | dim1_stride lo
    g1[7] = 0;                                                 // dim1_stride hi
    i32x4 gz = {0, 0, 0, 0};
#if __clang_major__ >= 23
    i32x8 gz8 = {0, 0, 0, 0, 0, 0, 0, 0};
    __builtin_amdgcn_tensor_load_to_lds(g0, g1, gz, gz, gz8, 0);
#else
    __builtin_amdgcn_tensor_load_to_lds(g0, g1, gz, gz, 0);
#endif
}

static __device__ __forceinline__ void wait_tensorcnt_0() {
#if __has_builtin(__builtin_amdgcn_s_wait_tensorcnt)
    __builtin_amdgcn_s_wait_tensorcnt(0);
#else
    asm volatile("s_wait_tensorcnt 0x0" ::: "memory");
#endif
}
static __device__ __forceinline__ void wait_tensorcnt_2() {
#if __has_builtin(__builtin_amdgcn_s_wait_tensorcnt)
    __builtin_amdgcn_s_wait_tensorcnt(2);
#else
    asm volatile("s_wait_tensorcnt 0x2" ::: "memory");
#endif
}
#endif // HAVE_TDM

// ---------------------------------------------------------------------------
// fp32 -> bf16 elementwise convert
// ---------------------------------------------------------------------------
__global__ __launch_bounds__(256)
void f32_to_bf16(const float* __restrict__ in, __bf16* __restrict__ out, int n) {
    int i = blockIdx.x * 256 + threadIdx.x;
    if (i < n) out[i] = (__bf16)in[i];
}

// ---------------------------------------------------------------------------
// C[M][N] = A[M][K] * W[N][K]^T   (bf16 inputs, f32 accumulate/output)
// block: 256 threads = 8 waves; block tile 128(M) x 128(N); K step 32.
// wave (wm in 0..1, wn in 0..3) computes 64x32 via 4x2 WMMA accumulators.
// LDS tiles fed by TDM (tensor_load_to_lds) with double buffering; wave 0
// issues the DMA, everyone synchronizes on a barrier after s_wait_tensorcnt.
// ---------------------------------------------------------------------------
__global__ __launch_bounds__(256)
void gemm_bf16_wmma(const __bf16* __restrict__ A, const __bf16* __restrict__ W,
                    float* __restrict__ C, int M, int N, int K)
{
    __shared__ __bf16 As[2][128 * LDSW_];
    __shared__ __bf16 Bs[2][128 * LDSW_];

    const int tid  = threadIdx.x;
    const int lane = tid & 31;
    const int wave = tid >> 5;
    const int nle  = lane & 15;
    const int hi   = lane >> 4;
    const int wm   = wave & 1;
    const int wn   = wave >> 1;
    const int m0   = blockIdx.y * 128;
    const int n0   = blockIdx.x * 128;

    const v8f z8 = {0.f,0.f,0.f,0.f,0.f,0.f,0.f,0.f};
    v8f acc[4][2];
    #pragma unroll
    for (int mt = 0; mt < 4; ++mt)
        #pragma unroll
        for (int nt = 0; nt < 2; ++nt) acc[mt][nt] = z8;

    const __bf16* aTile = A + (size_t)m0 * K;
    const __bf16* bTile = W + (size_t)n0 * K;

#ifdef HAVE_TDM
    // ISA §10.2: flat->LDS mapping truncates to addr[31:0]
    const unsigned ldsA[2] = { (unsigned)(size_t)&As[0][0], (unsigned)(size_t)&As[1][0] };
    const unsigned ldsB[2] = { (unsigned)(size_t)&Bs[0][0], (unsigned)(size_t)&Bs[1][0] };

    if (tid == 0) {                    // TDM ignores EXEC; one wave issues
        tdm_load_tile_128x32(aTile, ldsA[0], M, K);
        tdm_load_tile_128x32(bTile, ldsB[0], N, K);
    }
#else
    const int srow = tid >> 1;
    const int scol = (tid & 1) << 4;
    const __bf16* ag = aTile + (size_t)srow * K + scol;
    const __bf16* bg = bTile + (size_t)srow * K + scol;
#endif

    for (int k0 = 0; k0 < K; k0 += 32) {
        const int buf = (k0 >> 5) & 1;

#ifdef HAVE_TDM
        if (tid == 0) {
            if (k0 + 32 < K) {
                tdm_load_tile_128x32(aTile + k0 + 32, ldsA[buf ^ 1], M, K);
                tdm_load_tile_128x32(bTile + k0 + 32, ldsB[buf ^ 1], N, K);
                wait_tensorcnt_2();    // current buffer's two tiles have landed
            } else {
                wait_tensorcnt_0();
            }
        }
        __syncthreads();
#else
        *(v8bf*)(&As[0][0] + srow * LDSW_ + scol)     = *(const v8bf*)(ag + k0);
        *(v8bf*)(&As[0][0] + srow * LDSW_ + scol + 8) = *(const v8bf*)(ag + k0 + 8);
        *(v8bf*)(&Bs[0][0] + srow * LDSW_ + scol)     = *(const v8bf*)(bg + k0);
        *(v8bf*)(&Bs[0][0] + srow * LDSW_ + scol + 8) = *(const v8bf*)(bg + k0 + 8);
        __syncthreads();
#endif

#ifdef HAVE_TDM
        const __bf16* aBuf = &As[buf][0];
        const __bf16* bBuf = &Bs[buf][0];
#else
        const __bf16* aBuf = &As[0][0];
        const __bf16* bBuf = &Bs[0][0];
#endif
        v16bf bf[2];
        #pragma unroll
        for (int nt = 0; nt < 2; ++nt) {
            const __bf16* bp = bBuf + (wn * 32 + nt * 16 + nle) * LDSW_ + hi * 16;
            bf[nt] = cat8(*(const v8bf*)bp, *(const v8bf*)(bp + 8));
        }
        #pragma unroll
        for (int mt = 0; mt < 4; ++mt) {
            const __bf16* ap = aBuf + (wm * 64 + mt * 16 + nle) * LDSW_;
            v16bf af = cat8(*(const v8bf*)(ap + hi * 16),
                            *(const v8bf*)(ap + hi * 16 + 8));
            #pragma unroll
            for (int nt = 0; nt < 2; ++nt)
                acc[mt][nt] = wmma_bf16(af, bf[nt], acc[mt][nt]);
        }
        __syncthreads();
    }

    #pragma unroll
    for (int mt = 0; mt < 4; ++mt)
        #pragma unroll
        for (int nt = 0; nt < 2; ++nt)
            #pragma unroll
            for (int r = 0; r < 8; ++r) {
                int row = m0 + wm * 64 + mt * 16 + hi * 8 + r;
                int col = n0 + wn * 32 + nt * 16 + nle;
                C[(size_t)row * N + col] = acc[mt][nt][r];
            }
}

// ---------------------------------------------------------------------------
// RoPE + repack: QKV f32 [B][S][3][NH][HD] ->
//   Qb,Kb bf16 [B][NH][S][HD] (rotated), Vt bf16 [B][NH][HD][SP_] (transposed)
// positions are arange(S) so the int64 position_ids input is not needed.
// one thread per (b, s, h, d<32) rotary pair.
// ---------------------------------------------------------------------------
__global__ __launch_bounds__(256)
void rope_pack(const float* __restrict__ qkv, __bf16* __restrict__ Qb,
               __bf16* __restrict__ Kb, __bf16* __restrict__ Vt)
{
    int gid = blockIdx.x * 256 + threadIdx.x;
    int d = gid & 31;
    int h = (gid >> 5) % NH_;
    int s = ((gid >> 5) / NH_) % S_;
    int b = gid / (32 * NH_ * S_);

    const float* base = qkv + ((size_t)b * S_ + s) * QKVN_ + h * HD_ + d;
    float qlo = base[0],         qhi = base[32];
    float klo = base[HID_],      khi = base[HID_ + 32];
    float vlo = base[2 * HID_],  vhi = base[2 * HID_ + 32];

    float inv_freq = __powf(10000.0f, -(float)d * (1.0f / 32.0f));
    float ang = (float)s * inv_freq;
    float sn, cs;
    __sincosf(ang, &sn, &cs);

    size_t qb = (((size_t)b * NH_ + h) * S_ + s) * HD_ + d;
    Qb[qb]      = (__bf16)(qlo * cs - qhi * sn);
    Qb[qb + 32] = (__bf16)(qhi * cs + qlo * sn);
    Kb[qb]      = (__bf16)(klo * cs - khi * sn);
    Kb[qb + 32] = (__bf16)(khi * cs + klo * sn);

    size_t vb = (((size_t)b * NH_ + h) * HD_ + d) * SP_ + 64 + s;
    Vt[vb]            = (__bf16)vlo;
    Vt[vb + 32 * SP_] = (__bf16)vhi;
}

// ---------------------------------------------------------------------------
// Sliding-window attention. One wave per 16-query tile; 4 waves per block.
// Keys: 10 tiles of 16 starting at q0-64 (covers band, extras masked).
// Scores: 20 WMMAs -> analytic band mask -> shfl softmax -> P to LDS (bf16)
// -> 20 PV WMMAs.  Output written bf16 [B][S][NH*HD] for the final GEMM.
// ---------------------------------------------------------------------------
__global__ __launch_bounds__(128)
void attn_sw_wmma(const __bf16* __restrict__ Qb, const __bf16* __restrict__ Kb,
                  const __bf16* __restrict__ Vt, __bf16* __restrict__ Ab)
{
    __shared__ __bf16 plds[4][16][160];

    const int tid  = threadIdx.x;
    const int lane = tid & 31;
    const int w    = tid >> 5;
    const int nle  = lane & 15;
    const int hi   = lane >> 4;

    const int tile = blockIdx.x * 4 + w;
    const int qt = tile & 127;               // S_/16 = 128
    const int h  = (tile >> 7) % NH_;
    const int b  = tile / (128 * NH_);
    const int q0 = qt * 16;
    const int key0 = q0 - WIN_;
    const size_t bh = (size_t)b * NH_ + h;

    // Q fragments (A-matrix), kept in registers for the whole tile
    v16bf aq[2];
    {
        const __bf16* qrow = Qb + (bh * S_ + q0 + nle) * HD_;
        #pragma unroll
        for (int c = 0; c < 2; ++c)
            aq[c] = cat8(*(const v8bf*)(qrow + c * 32 + hi * 16),
                         *(const v8bf*)(qrow + c * 32 + hi * 16 + 8));
    }

    const v8f z8 = {0.f,0.f,0.f,0.f,0.f,0.f,0.f,0.f};
    v8f sc[10];
    #pragma unroll
    for (int t = 0; t < 10; ++t) {
        v8f a = z8;
        int kr = key0 + t * 16 + nle;
        kr = kr < 0 ? 0 : (kr > S_ - 1 ? S_ - 1 : kr);   // clamped rows masked below
        const __bf16* krow = Kb + (bh * S_ + kr) * HD_;
        #pragma unroll
        for (int c = 0; c < 2; ++c) {
            v16bf bk = cat8(*(const v8bf*)(krow + c * 32 + hi * 16),
                            *(const v8bf*)(krow + c * 32 + hi * 16 + 8));
            a = wmma_bf16(aq[c], bk, a);
        }
        sc[t] = a;
    }

    // analytic band mask + scale (1/sqrt(64))
    #pragma unroll
    for (int t = 0; t < 10; ++t) {
        int j = key0 + t * 16 + nle;
        #pragma unroll
        for (int r = 0; r < 8; ++r) {
            int i  = q0 + hi * 8 + r;
            int dd = i - j;
            bool valid = (j >= 0) && (j < S_) && (dd <= WIN_) && (dd >= -WIN_);
            sc[t][r] = valid ? sc[t][r] * 0.125f : -1e30f;
        }
    }

    // softmax over 160 (row's columns live across 16 lanes x 10 tiles)
    float mx[8], inv[8];
    #pragma unroll
    for (int r = 0; r < 8; ++r) {
        float m = sc[0][r];
        #pragma unroll
        for (int t = 1; t < 10; ++t) m = fmaxf(m, sc[t][r]);
        #pragma unroll
        for (int x = 1; x < 16; x <<= 1) m = fmaxf(m, __shfl_xor(m, x, 32));
        mx[r] = m;
    }
    #pragma unroll
    for (int r = 0; r < 8; ++r) {
        float s = 0.f;
        #pragma unroll
        for (int t = 0; t < 10; ++t) {
            float p = __expf(sc[t][r] - mx[r]);
            sc[t][r] = p;
            s += p;
        }
        #pragma unroll
        for (int x = 1; x < 16; x <<= 1) s += __shfl_xor(s, x, 32);
        inv[r] = 1.0f / s;
    }

    // normalized probs -> LDS (C-layout -> A-layout redistribution)
    #pragma unroll
    for (int t = 0; t < 10; ++t)
        #pragma unroll
        for (int r = 0; r < 8; ++r)
            plds[w][hi * 8 + r][t * 16 + nle] = (__bf16)(sc[t][r] * inv[r]);
    __syncthreads();

    // attn = P(16x160) @ V(160x64); V transposed so B-fragments are contiguous
    v8f oacc[4];
    #pragma unroll
    for (int nt = 0; nt < 4; ++nt) oacc[nt] = z8;
    #pragma unroll
    for (int c = 0; c < 5; ++c) {
        v16bf ap = cat8(*(const v8bf*)&plds[w][nle][c * 32 + hi * 16],
                        *(const v8bf*)&plds[w][nle][c * 32 + hi * 16 + 8]);
        #pragma unroll
        for (int nt = 0; nt < 4; ++nt) {
            const __bf16* vrow = Vt + (bh * HD_ + nt * 16 + nle) * SP_
                               + (q0 + c * 32 + hi * 16);   // q0 = 64 + key0
            v16bf bv = cat8(*(const v8bf*)vrow, *(const v8bf*)(vrow + 8));
            oacc[nt] = wmma_bf16(ap, bv, oacc[nt]);
        }
    }

    #pragma unroll
    for (int nt = 0; nt < 4; ++nt)
        #pragma unroll
        for (int r = 0; r < 8; ++r)
            Ab[((size_t)b * S_ + q0 + hi * 8 + r) * HID_ + h * HD_ + nt * 16 + nle]
                = (__bf16)oacc[nt][r];
}

// ---------------------------------------------------------------------------
extern "C" void kernel_launch(void* const* d_in, const int* in_sizes, int n_in,
                              void* d_out, int out_size, void* d_ws, size_t ws_size,
                              hipStream_t stream)
{
    (void)in_sizes; (void)n_in; (void)out_size; (void)ws_size;
    const float* X    = (const float*)d_in[0];
    const float* Wqkv = (const float*)d_in[1];
    const float* Wo   = (const float*)d_in[2];
    // d_in[3] (mask) and d_in[4] (position_ids) are reproduced analytically.
    float* out = (float*)d_out;

    char* p = (char*)d_ws;
    auto take = [&](size_t bytes) -> void* {
        void* r = (void*)p;
        p += (bytes + 255) & ~(size_t)255;
        return r;
    };
    __bf16* Xb    = (__bf16*)take((size_t)B_ * S_ * HID_ * 2);
    __bf16* Wqkvb = (__bf16*)take((size_t)QKVN_ * HID_ * 2);
    __bf16* Wob   = (__bf16*)take((size_t)HID_ * HID_ * 2);
    float*  QKVf  = (float*) take((size_t)B_ * S_ * QKVN_ * 4);
    __bf16* Qb    = (__bf16*)take((size_t)B_ * NH_ * S_ * HD_ * 2);
    __bf16* Kb    = (__bf16*)take((size_t)B_ * NH_ * S_ * HD_ * 2);
    __bf16* Vt    = (__bf16*)take((size_t)B_ * NH_ * HD_ * SP_ * 2);
    __bf16* Ab    = (__bf16*)take((size_t)B_ * S_ * HID_ * 2);

    const int nX  = B_ * S_ * HID_;
    const int nWq = QKVN_ * HID_;
    const int nWo = HID_ * HID_;
    f32_to_bf16<<<(nX  + 255) / 256, 256, 0, stream>>>(X,    Xb,    nX);
    f32_to_bf16<<<(nWq + 255) / 256, 256, 0, stream>>>(Wqkv, Wqkvb, nWq);
    f32_to_bf16<<<(nWo + 255) / 256, 256, 0, stream>>>(Wo,   Wob,   nWo);

    // QKV projection: (4096 x 768) x (2304 x 768)^T
    gemm_bf16_wmma<<<dim3(QKVN_ / 128, (B_ * S_) / 128), 256, 0, stream>>>(
        Xb, Wqkvb, QKVf, B_ * S_, QKVN_, HID_);

    // zero V pad regions, then RoPE + repack
    hipMemsetAsync(Vt, 0, (size_t)B_ * NH_ * HD_ * SP_ * 2, stream);
    const int nR = B_ * S_ * NH_ * 32;
    rope_pack<<<nR / 256, 256, 0, stream>>>(QKVf, Qb, Kb, Vt);

    // sliding-window attention: 3072 query tiles, 4 per block
    attn_sw_wmma<<<(B_ * NH_ * (S_ / 16)) / 4, 128, 0, stream>>>(Qb, Kb, Vt, Ab);

    // output projection: (4096 x 768) x (768 x 768)^T -> d_out (f32)
    gemm_bf16_wmma<<<dim3(HID_ / 128, (B_ * S_) / 128), 256, 0, stream>>>(
        Ab, Wob, out, B_ * S_, HID_, HID_);
}
